// MessageLayer_82875688944111
// MI455X (gfx1250) — compile-verified, matrix operand order checked by reading, not run on hardware
//
#include <hip/hip_runtime.h>

typedef __attribute__((ext_vector_type(16))) _Float16 v16h;
typedef __attribute__((ext_vector_type(8)))  _Float16 v8h;
typedef __attribute__((ext_vector_type(8)))  float    v8f;

#define FD 64   // feature dim D

// ---------------------------------------------------------------------------
// lower_bound over sorted int array
// ---------------------------------------------------------------------------
__device__ __forceinline__ int lower_bound_i(const int* __restrict__ a, int len, int key) {
    int lo = 0, hi = len;
    while (lo < hi) {
        int mid = (lo + hi) >> 1;
        if (a[mid] < key) lo = mid + 1; else hi = mid;
    }
    return lo;
}

// ---------------------------------------------------------------------------
// Pack W1 [128,64], W2 [64,64] (f32 row-major) and Wg [64,1] into WMMA
// B-fragment layout. frag(kt,nt) is 32x16 f16, per-lane 16 contiguous halves:
//   n = nt*16 + (lane&15); kBase = lane<16 ? 0 : 8
//   half h<8  -> K = kt*32 + kBase + h
//   half h>=8 -> K = kt*32 + kBase + 16 + (h-8)
// Wg is packed as 2 fragments (kt=0,1) with only column n==0 nonzero.
// ---------------------------------------------------------------------------
__global__ void pack_weights_kernel(const float* __restrict__ W1,
                                    const float* __restrict__ W2,
                                    const float* __restrict__ Wg,
                                    _Float16* __restrict__ pW1,
                                    _Float16* __restrict__ pW2,
                                    _Float16* __restrict__ pWg) {
    for (int i = threadIdx.x; i < 4 * 4 * 512; i += blockDim.x) {
        int frag = i >> 9, kt = frag >> 2, nt = frag & 3;
        int r = i & 511, lane = r >> 4, h = r & 15;
        int kBase = (lane < 16) ? 0 : 8;
        int k = kt * 32 + ((h < 8) ? (kBase + h) : (kBase + 16 + (h - 8)));
        int n = nt * 16 + (lane & 15);
        pW1[i] = (_Float16)W1[k * FD + n];
    }
    for (int i = threadIdx.x; i < 2 * 4 * 512; i += blockDim.x) {
        int frag = i >> 9, kt = frag >> 2, nt = frag & 3;
        int r = i & 511, lane = r >> 4, h = r & 15;
        int kBase = (lane < 16) ? 0 : 8;
        int k = kt * 32 + ((h < 8) ? (kBase + h) : (kBase + 16 + (h - 8)));
        int n = nt * 16 + (lane & 15);
        pW2[i] = (_Float16)W2[k * FD + n];
    }
    for (int i = threadIdx.x; i < 2 * 512; i += blockDim.x) {
        int kt = i >> 9;
        int r = i & 511, lane = r >> 4, h = r & 15;
        int kBase = (lane < 16) ? 0 : 8;
        int k = kt * 32 + ((h < 8) ? (kBase + h) : (kBase + 16 + (h - 8)));
        int n = lane & 15;
        pWg[i] = (n == 0) ? (_Float16)Wg[k] : (_Float16)0.0f;
    }
}

// ---------------------------------------------------------------------------
// Edge MLP: one wave per 16-edge tile.
//   A1 [16x128]: gather-concat(atom_fea[self], atom_fea[nbr]) -> f16
//   H = leakyrelu(A1 @ W1 + b1); O = H @ W2 + b2
//   fea16[e,:] = O;  gate[e] = O . Wg + bg   (gate via 2 extra WMMAs)
// 26 x v_wmma_f32_16x16x32_f16 per tile.
// ---------------------------------------------------------------------------
__global__ __launch_bounds__(256) void edge_mlp_kernel(
    const float* __restrict__ atom_fea,   // [N,64]
    const int*   __restrict__ self_idx,   // [M]
    const int*   __restrict__ nbr_idx,    // [M]
    const float* __restrict__ b1,         // [64]
    const float* __restrict__ b2,         // [64]
    const float* __restrict__ bg,         // [1]
    const _Float16* __restrict__ pW1,     // packed [4][4][32][16]
    const _Float16* __restrict__ pW2,     // packed [2][4][32][16]
    const _Float16* __restrict__ pWg,     // packed [2][32][16]
    _Float16* __restrict__ fea16,         // [M,64]
    float*    __restrict__ gate,          // [M]
    int M_)
{
    __shared__ __align__(32) _Float16 sW1[4 * 4 * 512];  // 16 KB
    __shared__ __align__(32) _Float16 sW2[2 * 4 * 512];  //  8 KB
    __shared__ __align__(32) _Float16 sWg[2 * 512];      //  2 KB
    __shared__ __align__(32) _Float16 sH[8][16 * FD];    // 16 KB (per-wave 16x64 tile)

    for (int i = threadIdx.x; i < 4 * 4 * 512; i += 256) sW1[i] = pW1[i];
    for (int i = threadIdx.x; i < 2 * 4 * 512; i += 256) sW2[i] = pW2[i];
    for (int i = threadIdx.x; i < 2 * 512; i += 256)     sWg[i] = pWg[i];
    __syncthreads();

    const int wave = threadIdx.x >> 5;
    const int lane = threadIdx.x & 31;
    const int tile = blockIdx.x * 8 + wave;
    const long e0  = (long)tile * 16;
    if (e0 >= (long)M_) return;            // wave-uniform exit (EXEC stays full for WMMA)

    const int mrow  = lane & 15;           // row (edge) for A / col (feature) for B,C,D
    const int kBase = (lane < 16) ? 0 : 8;
    const int mOff  = (lane < 16) ? 0 : 8;

    long e = e0 + mrow; if (e >= (long)M_) e = (long)M_ - 1;   // clamp tail rows
    const float* pS = atom_fea + (long)self_idx[e] * FD;
    const float* pN = atom_fea + (long)nbr_idx[e]  * FD;
    _Float16* myH = &sH[wave][0];

    // ---- layer 1: [16x128] x [128x64] ----
    v8f acc1[4];
#pragma unroll
    for (int nt = 0; nt < 4; ++nt) {
        float bv = b1[nt * 16 + mrow];
#pragma unroll
        for (int r = 0; r < 8; ++r) acc1[nt][r] = bv;
    }
#pragma unroll
    for (int kt = 0; kt < 4; ++kt) {
        const float* src = (kt < 2) ? pS : pN;
        int f0 = (kt & 1) * 32 + kBase;
        float4 r0 = *(const float4*)(src + f0);
        float4 r1 = *(const float4*)(src + f0 + 4);
        float4 r2 = *(const float4*)(src + f0 + 16);
        float4 r3 = *(const float4*)(src + f0 + 20);
        v16h a;
        a[0]=(_Float16)r0.x; a[1]=(_Float16)r0.y; a[2]=(_Float16)r0.z; a[3]=(_Float16)r0.w;
        a[4]=(_Float16)r1.x; a[5]=(_Float16)r1.y; a[6]=(_Float16)r1.z; a[7]=(_Float16)r1.w;
        a[8]=(_Float16)r2.x; a[9]=(_Float16)r2.y; a[10]=(_Float16)r2.z; a[11]=(_Float16)r2.w;
        a[12]=(_Float16)r3.x; a[13]=(_Float16)r3.y; a[14]=(_Float16)r3.z; a[15]=(_Float16)r3.w;
#pragma unroll
        for (int nt = 0; nt < 4; ++nt) {
            v16h b = *(const v16h*)&sW1[(kt * 4 + nt) * 512 + lane * 16];
            acc1[nt] = __builtin_amdgcn_wmma_f32_16x16x32_f16(
                false, a, false, b, (short)0, acc1[nt], false, false);
        }
    }

    // ---- LeakyReLU(0.01), relayout D->A via per-wave LDS tile (f16) ----
#pragma unroll
    for (int nt = 0; nt < 4; ++nt) {
        int col = nt * 16 + mrow;
#pragma unroll
        for (int r = 0; r < 8; ++r) {
            float v = acc1[nt][r];
            v = (v > 0.0f) ? v : 0.01f * v;
            myH[(r + mOff) * FD + col] = (_Float16)v;
        }
    }

    // ---- layer 2: [16x64] x [64x64] ----
    v8f acc2[4];
#pragma unroll
    for (int nt = 0; nt < 4; ++nt) {
        float bv = b2[nt * 16 + mrow];
#pragma unroll
        for (int r = 0; r < 8; ++r) acc2[nt][r] = bv;
    }
#pragma unroll
    for (int kt = 0; kt < 2; ++kt) {
        int f0 = kt * 32 + kBase;
        v8h lo = *(const v8h*)&myH[mrow * FD + f0];
        v8h hi = *(const v8h*)&myH[mrow * FD + f0 + 16];
        v16h a;
#pragma unroll
        for (int i = 0; i < 8; ++i) { a[i] = lo[i]; a[8 + i] = hi[i]; }
#pragma unroll
        for (int nt = 0; nt < 4; ++nt) {
            v16h b = *(const v16h*)&sW2[(kt * 4 + nt) * 512 + lane * 16];
            acc2[nt] = __builtin_amdgcn_wmma_f32_16x16x32_f16(
                false, a, false, b, (short)0, acc2[nt], false, false);
        }
    }

    // ---- write O back into LDS tile (f16), overwriting H ----
#pragma unroll
    for (int nt = 0; nt < 4; ++nt) {
        int col = nt * 16 + mrow;
#pragma unroll
        for (int r = 0; r < 8; ++r)
            myH[(r + mOff) * FD + col] = (_Float16)acc2[nt][r];
    }

    // ---- gate = O . Wg via 2 WMMAs (B has Wg in column 0) ----
    v8f accg;
#pragma unroll
    for (int r = 0; r < 8; ++r) accg[r] = 0.0f;
#pragma unroll
    for (int kt = 0; kt < 2; ++kt) {
        int f0 = kt * 32 + kBase;
        v8h lo = *(const v8h*)&myH[mrow * FD + f0];
        v8h hi = *(const v8h*)&myH[mrow * FD + f0 + 16];
        v16h a;
#pragma unroll
        for (int i = 0; i < 8; ++i) { a[i] = lo[i]; a[8 + i] = hi[i]; }
        v16h b = *(const v16h*)&sWg[kt * 512 + lane * 16];
        accg = __builtin_amdgcn_wmma_f32_16x16x32_f16(
            false, a, false, b, (short)0, accg, false, false);
    }
    // column 0 lives on lanes 0 (rows 0..7) and 16 (rows 8..15)
    if ((lane & 15) == 0) {
        float bgv = bg[0];
#pragma unroll
        for (int r = 0; r < 8; ++r) {
            long ee = e0 + r + mOff;
            if (ee < (long)M_) gate[ee] = accg[r] + bgv;
        }
    }

    // ---- stream O rows to global fea16 (each lane: 32 contiguous halves) ----
    {
        int rr = lane >> 1;
        long ee = e0 + rr;
        if (ee < (long)M_) {
            int ch = (lane & 1) * 32;
            const v8h* s = (const v8h*)&myH[rr * FD + ch];
            v8h x0 = s[0], x1 = s[1], x2 = s[2], x3 = s[3];
            v8h* d = (v8h*)&fea16[ee * FD + ch];
            d[0] = x0; d[1] = x1; d[2] = x2; d[3] = x3;
        }
    }
}

// ---------------------------------------------------------------------------
// Per-node (sorted segments): gmax, then w[e] = aw[nbr]*exp(g-gmax) in place,
// denom[n] = sum w. Deterministic, no atomics.
// ---------------------------------------------------------------------------
__global__ void node_gate_kernel(const int* __restrict__ self_idx,
                                 const int* __restrict__ nbr_idx,
                                 const float* __restrict__ aw,      // [N,1]
                                 float* __restrict__ gate_w,        // [M] in/out
                                 float* __restrict__ denom,         // [N]
                                 int N_, int M_) {
    int n = blockIdx.x * blockDim.x + threadIdx.x;
    if (n >= N_) return;
    int lo = lower_bound_i(self_idx, M_, n);
    int hi = lower_bound_i(self_idx, M_, n + 1);
    float gm = -3.4e38f;
    for (int e = lo; e < hi; ++e) gm = fmaxf(gm, gate_w[e]);
    float dsum = 0.0f;
    for (int e = lo; e < hi; ++e) {
        float w = aw[nbr_idx[e]] * __expf(gate_w[e] - gm);
        gate_w[e] = w;
        dsum += w;
    }
    denom[n] = dsum;
}

// ---------------------------------------------------------------------------
// Pool: thread per (node, feature). out = residual + sum w/denom * fea16.
// Consecutive threads -> consecutive features -> coalesced fea16 row reads.
// ---------------------------------------------------------------------------
__global__ __launch_bounds__(256) void node_pool_kernel(
    const int* __restrict__ self_idx,
    const float* __restrict__ atom_fea,
    const _Float16* __restrict__ fea16,
    const float* __restrict__ w,
    const float* __restrict__ denom,
    float* __restrict__ out,
    int N_, int M_) {
    long gid = (long)blockIdx.x * blockDim.x + threadIdx.x;
    if (gid >= (long)N_ * FD) return;
    int n = (int)(gid >> 6);
    int f = (int)(gid & 63);
    int lo = lower_bound_i(self_idx, M_, n);
    int hi = lower_bound_i(self_idx, M_, n + 1);
    float inv = 1.0f / (denom[n] + 1e-13f);
    float acc = 0.0f;
    for (int e = lo; e < hi; ++e)
        acc += (w[e] * inv) * (float)fea16[(long)e * FD + f];
    out[gid] = atom_fea[gid] + acc;
}

// ---------------------------------------------------------------------------
extern "C" void kernel_launch(void* const* d_in, const int* in_sizes, int n_in,
                              void* d_out, int out_size, void* d_ws, size_t ws_size,
                              hipStream_t stream) {
    const float* aw   = (const float*)d_in[0];   // [N,1]
    const float* afea = (const float*)d_in[1];   // [N,64]
    const int*   self = (const int*)d_in[2];     // [M]
    const int*   nbr  = (const int*)d_in[3];     // [M]
    const float* W1   = (const float*)d_in[4];   // [128,64]
    const float* b1   = (const float*)d_in[5];   // [64]
    const float* W2   = (const float*)d_in[6];   // [64,64]
    const float* b2   = (const float*)d_in[7];   // [64]
    const float* Wg   = (const float*)d_in[8];   // [64,1]
    const float* bg   = (const float*)d_in[9];   // [1]
    float* out = (float*)d_out;

    const int N_ = in_sizes[0];
    const int M_ = in_sizes[2];

    // workspace carve (256B aligned regions)
    char* ws = (char*)d_ws;
    size_t off = 0;
    auto carve = [&](size_t bytes) -> void* {
        off = (off + 255) & ~(size_t)255;
        void* p = ws + off;
        off += bytes;
        return p;
    };
    _Float16* pW1   = (_Float16*)carve((size_t)4 * 4 * 512 * sizeof(_Float16));
    _Float16* pW2   = (_Float16*)carve((size_t)2 * 4 * 512 * sizeof(_Float16));
    _Float16* pWg   = (_Float16*)carve((size_t)2 * 512 * sizeof(_Float16));
    _Float16* fea16 = (_Float16*)carve((size_t)M_ * FD * sizeof(_Float16));
    float*    wEdge = (float*)carve((size_t)M_ * sizeof(float));
    float*    denom = (float*)carve((size_t)N_ * sizeof(float));

    pack_weights_kernel<<<1, 256, 0, stream>>>(W1, W2, Wg, pW1, pW2, pWg);

    int tiles = (M_ + 15) / 16;
    edge_mlp_kernel<<<(tiles + 7) / 8, 256, 0, stream>>>(
        afea, self, nbr, b1, b2, bg, pW1, pW2, pWg, fea16, wEdge, M_);

    node_gate_kernel<<<(N_ + 255) / 256, 256, 0, stream>>>(
        self, nbr, aw, wEdge, denom, N_, M_);

    long tot = (long)N_ * FD;
    node_pool_kernel<<<(int)((tot + 255) / 256), 256, 0, stream>>>(
        self, afea, fea16, wEdge, denom, out, N_, M_);
}
